// DeformConvNet_16527034155495
// MI455X (gfx1250) — matrix operand order repaired and checked by reading, not compiled, not run on hardware
//
#include <hip/hip_runtime.h>

// ---------------- problem constants ----------------
#define BATCH 8
#define CHN   128
#define CD    64
#define HH    128
#define WW    128
#define HW    (HH*WW)          // 16384

typedef __attribute__((ext_vector_type(16))) __bf16 v16bf;
typedef __attribute__((ext_vector_type(4)))  __bf16 v4bf;
typedef __attribute__((ext_vector_type(2)))  __bf16 v2bf;
typedef __attribute__((ext_vector_type(8)))  float  v8f;

__device__ __forceinline__ v8f v8f_zero() {
  v8f z = {0.f,0.f,0.f,0.f,0.f,0.f,0.f,0.f};
  return z;
}

// ---- A fragment from pixel-major LDS tile ([row=M][K contiguous], stride ld).
// ISA 16x32 bf16 A layout: lane&15 = row M; lanes 0-15 hold K [0..7]+[16..23],
// lanes 16-31 hold K [8..15]+[24..31].
__device__ __forceinline__ v16bf load_a_frag(const __bf16* base, int ld) {
  const int lane = threadIdx.x & 31;
  const __bf16* p = base + (lane & 15) * ld + ((lane >> 4) << 3);
  v16bf f;
#pragma unroll
  for (int i = 0; i < 8; ++i) { f[i] = p[i]; f[8 + i] = p[16 + i]; }
  return f;
}

// ---- A fragment from K-major (transposed) LDS tile ([k][M contiguous], stride ld).
__device__ __forceinline__ v16bf load_a_frag_T(const __bf16* base, int ld) {
  const int lane = threadIdx.x & 31;
  const int m = lane & 15;
  const int h8 = (lane >> 4) << 3;
  v16bf f;
#pragma unroll
  for (int i = 0; i < 8; ++i) {
    f[i]     = base[(h8 + i) * ld + m];
    f[8 + i] = base[(16 + h8 + i) * ld + m];
  }
  return f;
}

// ---- B fragment: 32x16 bf16 (KxN), memory layout [N][K] K-contiguous (stride ld).
// lanes 0-15: K [0..7]+[16..23] of column N=lane; lanes 16-31: K [8..15]+[24..31] of column N=lane-16.
__device__ __forceinline__ v16bf load_b_frag(const __bf16* base, int ld) {
  const int lane = threadIdx.x & 31;
  const __bf16* p = base + (lane & 15) * ld + ((lane >> 4) << 3);
  v16bf f;
#pragma unroll
  for (int i = 0; i < 8; ++i) { f[i] = p[i]; f[8 + i] = p[16 + i]; }
  return f;
}

__device__ __forceinline__ v8f wmma_bf16(v16bf a, v16bf b, v8f c) {
  return __builtin_amdgcn_wmma_f32_16x16x32_bf16(false, a, false, b, (short)0, c,
                                                 false, false);
}

__device__ __forceinline__ float mishf(float v) {
  float sp = (v > 20.0f) ? v : log1pf(expf(v));
  return v * tanhf(sp);
}

// ---------------- weight prep: fp32 -> bf16, [N][K] K-contiguous ----------------
__global__ __launch_bounds__(256)
void k_prep(const float* __restrict__ w0, const float* __restrict__ w_off,
            const float* __restrict__ w3d, const float* __restrict__ wl,
            __bf16* __restrict__ w0T, __bf16* __restrict__ woffT,
            __bf16* __restrict__ w3dT, __bf16* __restrict__ wlT) {
  int t = blockIdx.x * blockDim.x + threadIdx.x;
  if (t < 64 * 128) {                       // w0T[o][c] = w0[c][o]   (64 x 128)
    int o = t >> 7, c = t & 127;
    w0T[t] = (__bf16)w0[c * 64 + o];
  }
  if (t < 9 * 9 * 128 * 64) {               // woffT[i][tap][o][c] = w_off[i][o][c][tap]
    int c = t & 63;
    int o = (t >> 6) & 127;
    int rest = t >> 13;
    int tap = rest % 9;
    int i = rest / 9;
    woffT[t] = (__bf16)w_off[(((size_t)i * 128 + o) * 64 + c) * 9 + tap];
  }
  if (t < 9 * 64 * 64) {                    // w3dT[i][o][c] = w3d[o][c][i]
    int c = t & 63;
    int o = (t >> 6) & 63;
    int i = t >> 12;
    w3dT[t] = (__bf16)w3d[(o * 64 + c) * 9 + i];
  }
  if (t < 128 * 192) {                      // wlT[o][c] = wl[c][o]  (128 x 192)
    int c = t % 192;
    int o = t / 192;
    wlT[t] = (__bf16)wl[c * 128 + o];
  }
}

// ---------------- c0: z = mish(x . w0 * s0 + b0) ----------------
// GEMM M=B*HW, K=128, N=64. 128 pixels per block, 8 waves x 16 px x 64 N.
// A tile staged K-major in LDS: float4 pixel chunk -> one packed 8B ds store.
__global__ __launch_bounds__(256)
void k_c0(const float* __restrict__ x, const __bf16* __restrict__ w0T,
          const float* __restrict__ s0, const float* __restrict__ b0,
          float* __restrict__ zf, __bf16* __restrict__ zb) {
  __shared__ __align__(16) __bf16 sA[32 * 132];   // [k][pixel], stride 132
  const int t = threadIdx.x, wv = t >> 5, lane = t & 31;
  const int p0 = blockIdx.x * 128;
  const int b = p0 >> 14, pimg0 = p0 & (HW - 1);
  v8f acc[4];
#pragma unroll
  for (int nt = 0; nt < 4; ++nt) acc[nt] = v8f_zero();

  for (int k0 = 0; k0 < 128; k0 += 32) {
#pragma unroll
    for (int j = 0; j < 4; ++j) {           // 1024 chunks: 32 ch x 32 pixel-groups
      int idx = t + 256 * j;
      int rg = (idx & 31) << 2;             // pixel group of 4
      int kk = idx >> 5;                    // channel 0..31
      const float* gp = &x[(size_t)(b * CHN + k0 + kk) * HW + pimg0 + rg];
      if (k0 + 32 < 128) __builtin_prefetch(gp + 32 * HW, 0, 1);
      float4 v4 = *(const float4*)gp;
      v4bf pk = {(__bf16)v4.x, (__bf16)v4.y, (__bf16)v4.z, (__bf16)v4.w};
      *(v4bf*)&sA[kk * 132 + rg] = pk;      // single b64 LDS store
    }
    __syncthreads();
    v16bf a = load_a_frag_T(&sA[wv * 16], 132);
#pragma unroll
    for (int nt = 0; nt < 4; ++nt)
      acc[nt] = wmma_bf16(a, load_b_frag(w0T + (nt * 16) * 128 + k0, 128), acc[nt]);
    __syncthreads();
  }
  const int half = lane >> 4;
#pragma unroll
  for (int nt = 0; nt < 4; ++nt) {
    int n = nt * 16 + (lane & 15);
    float sc = s0[n], bi = b0[n];
    float r[8];
#pragma unroll
    for (int v = 0; v < 8; ++v) r[v] = mishf(acc[nt][v] * sc + bi);
    int pimg = pimg0 + wv * 16 + (half << 3);        // 8 consecutive pixels per lane
    float* zp = &zf[(size_t)(b * CD + n) * HW + pimg];
    *(float4*)zp       = make_float4(r[0], r[1], r[2], r[3]);
    *(float4*)(zp + 4) = make_float4(r[4], r[5], r[6], r[7]);
    __bf16* zbp = &zb[(size_t)(b * HW + pimg) * CD + n];
#pragma unroll
    for (int v = 0; v < 8; ++v) zbp[(size_t)v * CD] = (__bf16)r[v];
  }
}

// ---------------- offset conv: off = conv3x3(z, w_off[branch]) ----------------
// One (b,h) row per block. Row of z (bf16 NHWC) staged via ASYNC load-to-LDS.
// LDS stride 72 elems (144B = 9*16B) keeps every 8-channel chunk 16B aligned.
__global__ __launch_bounds__(256)
void k_offconv(const __bf16* __restrict__ zb, const __bf16* __restrict__ woffT,
               float* __restrict__ off, int branch) {
  __shared__ __align__(16) __bf16 sZ[130 * 72];
  const int t = threadIdx.x, wv = t >> 5, lane = t & 31;
  const int b = blockIdx.x >> 7, h = blockIdx.x & 127;
  v8f acc[8];
#pragma unroll
  for (int nt = 0; nt < 8; ++nt) acc[nt] = v8f_zero();

  for (int dy = 0; dy < 3; ++dy) {
    int hs = h + dy - 1;
    if ((unsigned)hs < 128u) {
      if (t < 128) {                         // zero-pad positions -1 and 128
        int c = t & 63;
        int pos = (t >> 6) ? 129 : 0;
        sZ[pos * 72 + c] = (__bf16)0.f;
      }
      const __bf16* rowp = zb + (size_t)(b * HW + hs * 128) * CD;
      unsigned sbase = (unsigned)(unsigned long long)&sZ[0];
#pragma unroll
      for (int j = 0; j < 4; ++j) {          // 1024 async 16B chunks: 128 pos x 8 ch-groups
        int c = t + 256 * j;
        int pos = (c >> 3) + 1;
        int k8 = (c & 7) << 3;
        unsigned ldst = sbase + (unsigned)(pos * 72 + k8) * 2u;
        unsigned long long ga =
            (unsigned long long)(rowp + (size_t)(pos - 1) * CD + k8);
        asm volatile("global_load_async_to_lds_b128 %0, %1, off"
                     :: "v"(ldst), "v"(ga) : "memory");
      }
      asm volatile("s_wait_asynccnt 0x0" ::: "memory");
    } else {
#pragma unroll
      for (int j = 0; j < 37; ++j) {
        int idx = t + 256 * j;
        if (idx < 130 * 72) sZ[idx] = (__bf16)0.f;
      }
    }
    __syncthreads();
    for (int dx = 0; dx < 3; ++dx) {
      int tap = dy * 3 + dx;
#pragma unroll
      for (int k0 = 0; k0 < 64; k0 += 32) {
        v16bf a = load_a_frag(&sZ[(wv * 16 + dx) * 72 + k0], 72);
#pragma unroll
        for (int nt = 0; nt < 8; ++nt)
          acc[nt] = wmma_bf16(
              a,
              load_b_frag(woffT + ((size_t)((branch * 9 + tap) * 128 + nt * 16)) * 64 + k0, 64),
              acc[nt]);
      }
    }
    __syncthreads();
  }
#pragma unroll
  for (int nt = 0; nt < 8; ++nt) {
    int n = nt * 16 + (lane & 15);
#pragma unroll
    for (int v = 0; v < 8; ++v) {
      int m = v + ((lane >> 4) << 3);
      int px = wv * 16 + m;
      off[((size_t)(b * HW + h * 128 + px)) * 128 + n] = acc[nt][v];
    }
  }
}

// ---------------- bilinear sample of one channel ----------------
__device__ __forceinline__ float bilin(const float* __restrict__ zf,
                                       const float* __restrict__ ob,
                                       int b, int u, int hs, int wq,
                                       int p, int hi) {
  int ch = 2 * u + hi;
  float orow = ob[(size_t)p * 128 + ch];
  float ocol = ob[(size_t)(p + 1) * 128 + ch];
  float cy = fminf(fmaxf(orow + (float)hs, 0.f), 127.f);
  float cx = fminf(fmaxf(ocol + (float)wq, 0.f), 127.f);
  float fy = floorf(cy), fx = floorf(cx);
  float wy = cy - fy, wx = cx - fx;
  int y0i = (int)fy, x0i = (int)fx;
  int y1i = (int)ceilf(cy), x1i = (int)ceilf(cx);
  const float* plane = zf + (size_t)(b * CD + u) * HW;
  float v_lt = plane[y0i * 128 + x0i];
  float v_rt = plane[y1i * 128 + x0i];
  float v_lb = plane[y0i * 128 + x1i];
  float v_rb = plane[y1i * 128 + x1i];
  float top = v_lt + (v_rt - v_lt) * wy;
  float bot = v_lb + (v_rb - v_lb) * wy;
  return top + (bot - top) * wx;
}

// ---------------- sample + Conv3d-slice accumulate for one branch ----------------
// y kept NCHW f32 so the RMW epilogue and k_final staging vectorize to b128.
__global__ __launch_bounds__(256)
void k_sample(const float* __restrict__ zf, const float* __restrict__ off,
              const __bf16* __restrict__ w3dT, float* __restrict__ y,
              int branch, int dy, int dx, int first) {
  __shared__ __align__(16) __bf16 sS[130 * 66];
  const int t = threadIdx.x, wv = t >> 5, lane = t & 31;
  const int b = blockIdx.x >> 7, h = blockIdx.x & 127;
  const int hs = h + dy;
  const bool valid = (unsigned)hs < 128u;

  for (int j = 0; j < 17; ++j) {            // 130 pos x 32 channel-pairs
    int idx = t + 256 * j;
    if (idx < 130 * 32) {
      int u2 = (idx & 31) << 1;
      int pos = idx >> 5, wq = pos - 1;
      float sa = 0.f, sb = 0.f;
      if (valid && (unsigned)wq < 128u) {
        int pp = hs * 128 + wq;
        int q = 2 * pp;
        int hi = q >> 14;
        int p = q & (HW - 1);
        const float* ob = off + (size_t)(b * HW) * 128;
        sa = bilin(zf, ob, b, u2, hs, wq, p, hi);
        sb = bilin(zf, ob, b, u2 + 1, hs, wq, p, hi);
      }
      v2bf pk = {(__bf16)sa, (__bf16)sb};
      *(v2bf*)&sS[pos * 66 + u2] = pk;      // single b32 LDS store
    }
  }
  __syncthreads();

  v8f acc[4];
#pragma unroll
  for (int nt = 0; nt < 4; ++nt) acc[nt] = v8f_zero();
#pragma unroll
  for (int k0 = 0; k0 < 64; k0 += 32) {
    v16bf a = load_a_frag(&sS[(wv * 16 + dx + 1) * 66 + k0], 66);
#pragma unroll
    for (int nt = 0; nt < 4; ++nt)
      acc[nt] = wmma_bf16(a, load_b_frag(w3dT + (size_t)(branch * 64 + nt * 16) * 64 + k0, 64),
                          acc[nt]);
  }
  const int half = lane >> 4;
  const int px = wv * 16 + (half << 3);
#pragma unroll
  for (int nt = 0; nt < 4; ++nt) {
    int n = nt * 16 + (lane & 15);
    float* yp = &y[(size_t)(b * CD + n) * HW + h * 128 + px];
    float4 lo, hi;
    if (first) {
      lo = make_float4(0.f, 0.f, 0.f, 0.f);
      hi = lo;
    } else {
      lo = *(float4*)yp;
      hi = *(float4*)(yp + 4);
    }
    lo.x += acc[nt][0]; lo.y += acc[nt][1]; lo.z += acc[nt][2]; lo.w += acc[nt][3];
    hi.x += acc[nt][4]; hi.y += acc[nt][5]; hi.z += acc[nt][6]; hi.w += acc[nt][7];
    *(float4*)yp       = lo;
    *(float4*)(yp + 4) = hi;
  }
}

// ---------------- final: out = mish(concat(x, y+b3d) . wl * sl + bl) ----------------
__global__ __launch_bounds__(256)
void k_final(const float* __restrict__ x, const float* __restrict__ y,
             const float* __restrict__ b3d, const __bf16* __restrict__ wlT,
             const float* __restrict__ sl, const float* __restrict__ bl,
             float* __restrict__ out) {
  __shared__ __align__(16) __bf16 sA[32 * 132];   // [k][pixel], stride 132
  const int t = threadIdx.x, wv = t >> 5, lane = t & 31;
  const int p0 = blockIdx.x * 128;
  const int b = p0 >> 14, pimg0 = p0 & (HW - 1);
  v8f acc[8];
#pragma unroll
  for (int nt = 0; nt < 8; ++nt) acc[nt] = v8f_zero();

  for (int k0 = 0; k0 < 192; k0 += 32) {
#pragma unroll
    for (int j = 0; j < 4; ++j) {
      int idx = t + 256 * j;
      int rg = (idx & 31) << 2;
      int kk = idx >> 5;
      int kg = k0 + kk;
      float4 v4;
      if (kg < 128) {
        v4 = *(const float4*)&x[(size_t)(b * CHN + kg) * HW + pimg0 + rg];
      } else {
        v4 = *(const float4*)&y[(size_t)(b * CD + (kg - 128)) * HW + pimg0 + rg];
        float bb = b3d[kg - 128];
        v4.x += bb; v4.y += bb; v4.z += bb; v4.w += bb;
      }
      v4bf pk = {(__bf16)v4.x, (__bf16)v4.y, (__bf16)v4.z, (__bf16)v4.w};
      *(v4bf*)&sA[kk * 132 + rg] = pk;
    }
    __syncthreads();
    v16bf a = load_a_frag_T(&sA[wv * 16], 132);
#pragma unroll
    for (int nt = 0; nt < 8; ++nt)
      acc[nt] = wmma_bf16(a, load_b_frag(wlT + (nt * 16) * 192 + k0, 192), acc[nt]);
    __syncthreads();
  }
  const int half = lane >> 4;
#pragma unroll
  for (int nt = 0; nt < 8; ++nt) {
    int n = nt * 16 + (lane & 15);
    float sc = sl[n], bi = bl[n];
    float r[8];
#pragma unroll
    for (int v = 0; v < 8; ++v) r[v] = mishf(acc[nt][v] * sc + bi);
    int pimg = pimg0 + wv * 16 + (half << 3);
    float* op = &out[(size_t)(b * CHN + n) * HW + pimg];
    *(float4*)op       = make_float4(r[0], r[1], r[2], r[3]);
    *(float4*)(op + 4) = make_float4(r[4], r[5], r[6], r[7]);
  }
}

// ---------------- host launch ----------------
extern "C" void kernel_launch(void* const* d_in, const int* in_sizes, int n_in,
                              void* d_out, int out_size, void* d_ws, size_t ws_size,
                              hipStream_t stream) {
  (void)in_sizes; (void)n_in; (void)out_size; (void)ws_size;
  const float* x     = (const float*)d_in[0];
  const float* w0    = (const float*)d_in[1];
  const float* s0    = (const float*)d_in[2];
  const float* b0    = (const float*)d_in[3];
  const float* w_off = (const float*)d_in[4];
  const float* w3d   = (const float*)d_in[5];
  const float* b3d   = (const float*)d_in[6];
  const float* wl    = (const float*)d_in[7];
  const float* sl    = (const float*)d_in[8];
  const float* bl    = (const float*)d_in[9];

  char* ws = (char*)d_ws;
  float*  zf    = (float*)(ws + 0);            // 33,554,432 B : z f32 NCHW
  __bf16* zb    = (__bf16*)(ws + 33554432);    // 16,777,216 B : z bf16 NHWC
  float*  off   = (float*)(ws + 50331648);     // 67,108,864 B : offsets (B,HW,128)
  float*  yacc  = (float*)(ws + 117440512);    // 33,554,432 B : y f32 NCHW
  __bf16* w0T   = (__bf16*)(ws + 150994944);   //     16,384 B
  __bf16* woffT = (__bf16*)(ws + 151011328);   //  1,327,104 B
  __bf16* w3dT  = (__bf16*)(ws + 152338432);   //     73,728 B
  __bf16* wlT   = (__bf16*)(ws + 152412160);   //     49,152 B
  float* outp = (float*)d_out;

  k_prep<<<2592, 256, 0, stream>>>(w0, w_off, w3d, wl, w0T, woffT, w3dT, wlT);
  k_c0<<<BATCH * HW / 128, 256, 0, stream>>>(x, w0T, s0, b0, zf, zb);
  for (int i = 0; i < 9; ++i) {
    k_offconv<<<BATCH * HH, 256, 0, stream>>>(zb, woffT, off, i);
    k_sample<<<BATCH * HH, 256, 0, stream>>>(zf, off, w3dT, yacc,
                                             i, i / 3 - 1, i % 3 - 1, i == 0 ? 1 : 0);
  }
  k_final<<<BATCH * HW / 128, 256, 0, stream>>>(x, yacc, b3d, wlT, sl, bl, outp);
}
